// Encoder_13709535609671
// MI455X (gfx1250) — matrix operand order repaired and checked by reading, not compiled, not run on hardware
//
#include <hip/hip_runtime.h>
#include <hip/hip_bf16.h>
#include <math.h>

#define DI __device__ __forceinline__

typedef __attribute__((ext_vector_type(16))) __bf16 v16bf;
typedef __attribute__((ext_vector_type(8)))  float  v8f;

union FragBF { v16bf v; uint4 q[2]; };

constexpr int   kN   = 100000;
constexpr int   kE   = 400000;
constexpr int   kG   = 2048;
constexpr int   kINF = 9;
constexpr int   kEMB = 10;
constexpr int   kH   = 10;
constexpr int   kC   = 64;
constexpr int   kNL  = 5;
constexpr int   kOUT = 512;
constexpr int   kLD  = 330;   // 10 + 5*64 : leading dim of concat feature matrix
constexpr int   kHP  = 640;   // H*C
constexpr float kEPS = 1e-5f;

DI unsigned short f2bf(float f) {
  unsigned u = __float_as_uint(f);
  u += 0x7FFFu + ((u >> 16) & 1u);          // round-to-nearest-even
  return (unsigned short)(u >> 16);
}
DI float gelu_exact(float x) { return 0.5f * x * (1.0f + erff(x * 0.7071067811865475f)); }
DI void atomicMaxF(float* addr, float val) {
  if (val >= 0.0f) atomicMax((int*)addr, __float_as_int(val));
  else             atomicMin((unsigned int*)addr, __float_as_uint(val));
}

// ---------------------------------------------------------------------------
// Tiled bf16 WMMA GEMM: C[M,Nc] = A[M,K] @ B[K,Nc] (+bias), fp32 in/out.
// Block = 256 threads (8 waves). Tile 64x128, K-step 32. K must be even.
// Wave w: row tile = (w&3)*16, col group = (w>>2)*64, 4 col tiles per wave.
// All staging addresses are incremental 32-bit offsets (buffers < 2^31 elems),
// branchless (select-to-zero-offset), register-first so all global loads are
// in flight before any LDS store. B fragments preloaded before the WMMA chain.
// ---------------------------------------------------------------------------
#define TM 64
#define TN 128
#define TK 32
#define SA 40   // LDS stride (halves): keeps all fragment reads 16B-aligned
#define SB 40

__launch_bounds__(256)
__global__ void gemm_bf16_wmma(const float* __restrict__ A, int lda,
                               const float* __restrict__ B, int ldb,
                               const float* __restrict__ bias,
                               float* __restrict__ C, int ldc,
                               int M, int K) {
  __shared__ unsigned short As[TM * SA];   // 64 rows x 32 k (row-major)
  __shared__ unsigned short Bs[TN * SB];   // 128 n  x 32 k (transposed)

  const int m0   = blockIdx.x * TM;
  const int n0   = blockIdx.y * TN;
  const int tid  = threadIdx.x;
  const int lane = tid & 31;
  const int wave = tid >> 5;
  const int m_off = (wave & 3) * 16;
  const int ngrp  = (wave >> 2) * 64;      // 0 or 64
  const int lr = lane & 15;
  const int lg = lane >> 4;

  // staging coordinates (fixed per thread)
  const int a_r = tid >> 4;                // 0..15, +16 per j (4 j's)
  const int a_c = (tid & 15) * 2;          // even k column, float2
  const int b_k = tid >> 6;                // 0..3, +4 per j (8 j's)
  const int b_n = (tid & 63) * 2;          // even n column, float2
  const unsigned ldb32 = (unsigned)(ldb * TK);

  // incremental 32-bit element offsets (clamped rows are loop-invariant)
  unsigned aoff[4];
  bool     arow_ok[4];
#pragma unroll
  for (int j = 0; j < 4; ++j) {
    int gr = m0 + a_r + j * 16;
    arow_ok[j] = gr < M;
    aoff[j] = (unsigned)((arow_ok[j] ? gr : 0) * lda + a_c);
  }
  unsigned boff[8];
#pragma unroll
  for (int j = 0; j < 8; ++j)
    boff[j] = (unsigned)((b_k + j * 4) * ldb + n0 + b_n);

  v8f acc[4] = {{}, {}, {}, {}};

  for (int k0 = 0; k0 < K; k0 += TK) {
    __syncthreads();

    // ---- gather A tile (64x32) as float2 pairs, branchless ----
    float2 av[4];
#pragma unroll
    for (int j = 0; j < 4; ++j) {
      bool ok = arow_ok[j] & ((k0 + a_c) < K);   // K even => pair all-in/out
      unsigned off = ok ? aoff[j] : 0u;
      float2 v = *(const float2*)&A[off];
      av[j].x = ok ? v.x : 0.0f;
      av[j].y = ok ? v.y : 0.0f;
      aoff[j] += TK;
    }
    // ---- gather B tile (32x128) as float2 pairs along n, branchless ----
    float2 bv[8];
#pragma unroll
    for (int j = 0; j < 8; ++j) {
      bool ok = (k0 + b_k + j * 4) < K;
      unsigned off = ok ? boff[j] : 0u;
      float2 v = *(const float2*)&B[off];
      bv[j].x = ok ? v.x : 0.0f;
      bv[j].y = ok ? v.y : 0.0f;
      boff[j] += ldb32;
    }
    // prefetch next k-slab (offsets already advanced)
    if (k0 + TK < K) {
      __builtin_prefetch(&A[aoff[0]], 0, 1);
      __builtin_prefetch(&B[boff[0]], 0, 1);
    }

    // ---- convert + store to LDS ----
#pragma unroll
    for (int j = 0; j < 4; ++j) {
      unsigned p = (unsigned)f2bf(av[j].x) | ((unsigned)f2bf(av[j].y) << 16);
      *(unsigned*)&As[(a_r + j * 16) * SA + a_c] = p;   // 4B-aligned
    }
#pragma unroll
    for (int j = 0; j < 8; ++j) {
      int kk = b_k + j * 4;
      Bs[b_n * SB + kk]       = f2bf(bv[j].x);
      Bs[(b_n + 1) * SB + kk] = f2bf(bv[j].y);
    }
    __syncthreads();

    // ---- fragments: A once, all 4 B fragments, then 4 back-to-back WMMA ----
    // A layout: lanes 0-15 hold K {0..15}, lanes 16-31 hold {16..31}
    FragBF a;
    a.q[0] = *(const uint4*)&As[(m_off + lr) * SA + lg * 16];
    a.q[1] = *(const uint4*)&As[(m_off + lr) * SA + lg * 16 + 8];
    FragBF b[4];
#pragma unroll
    for (int t = 0; t < 4; ++t) {
      const uint4* p = (const uint4*)&Bs[(ngrp + t * 16 + lr) * SB + lg * 16];
      b[t].q[0] = p[0];
      b[t].q[1] = p[1];
    }
#pragma unroll
    for (int t = 0; t < 4; ++t)
      acc[t] = __builtin_amdgcn_wmma_f32_16x16x32_bf16(false, a.v, false, b[t].v,
                                                       (short)0, acc[t], false, false);
  }

  // C/D layout: VGPR r -> M = (lane/16)*8 + r, N = lane%16
  const int lh = lane >> 4;
#pragma unroll
  for (int t = 0; t < 4; ++t) {
#pragma unroll
    for (int r = 0; r < 8; ++r) {
      int gm = m0 + m_off + lh * 8 + r;
      if (gm < M) {
        int gn = n0 + ngrp + t * 16 + lr;
        float v = acc[t][r];
        if (bias) v += bias[gn];
        C[(unsigned)(gm * ldc + gn)] = v;
      }
    }
  }
}

// ---------------------------------------------------------------------------
// Small helper kernels
// ---------------------------------------------------------------------------
__global__ void fill_f32(float* p, float v, long long n) {
  long long i = (long long)blockIdx.x * blockDim.x + threadIdx.x;
  if (i < n) p[i] = v;
}

__global__ void embed_kernel(const float* __restrict__ x, const float* __restrict__ w,
                             const float* __restrict__ b, float* __restrict__ h) {
  int i = blockIdx.x * blockDim.x + threadIdx.x;   // n*10 + j
  if (i >= kN * kEMB) return;
  int n = i / kEMB, j = i - n * kEMB;
  float s = b[j];
  for (int k = 0; k < kINF; ++k) s += x[n * kINF + k] * w[k * kEMB + j];
  h[(size_t)n * kLD + j] = s;
}

__global__ void deg_loop_scatter(const int* __restrict__ dst, const float* __restrict__ ea,
                                 float* __restrict__ deg, float* __restrict__ lsum) {
  int e = blockIdx.x * blockDim.x + threadIdx.x;
  if (e >= kE) return;
  int t = dst[e];
  atomicAdd(&deg[t], 1.0f);
  atomicAdd(&lsum[t * 3 + 0], ea[(size_t)e * 3 + 0]);
  atomicAdd(&lsum[t * 3 + 1], ea[(size_t)e * 3 + 1]);
  atomicAdd(&lsum[t * 3 + 2], ea[(size_t)e * 3 + 2]);
}

__global__ void loop_finalize(float* __restrict__ lattr, const float* __restrict__ deg) {
  int i = blockIdx.x * blockDim.x + threadIdx.x;   // n*3 + f
  if (i >= kN * 3) return;
  float d = deg[i / 3];
  lattr[i] = lattr[i] / fmaxf(d, 1.0f);
}

__global__ void edge_weight_kernel(const float* __restrict__ lin_e,
                                   const float* __restrict__ att_e,
                                   float* __restrict__ we) {
  int i = threadIdx.x;                    // f*kH + h, 30 outputs
  if (i >= 3 * kH) return;
  int f = i / kH, h = i - f * kH;
  float s = 0.0f;
  for (int c = 0; c < kC; ++c) s += lin_e[f * kHP + h * kC + c] * att_e[h * kC + c];
  we[i] = s;
}

__global__ void att_node_kernel(const float* __restrict__ hp,
                                const float* __restrict__ att_src,
                                const float* __restrict__ att_dst,
                                float* __restrict__ asrc, float* __restrict__ adst) {
  int i = blockIdx.x * blockDim.x + threadIdx.x;   // n*kH + h
  if (i >= kN * kH) return;
  int n = i / kH, h = i - n * kH;
  const float* row = hp + (size_t)n * kHP + h * kC;
  const float* ws  = att_src + h * kC;
  const float* wd  = att_dst + h * kC;
  float s = 0.0f, d = 0.0f;
  for (int c = 0; c < kC; ++c) { float v = row[c]; s += v * ws[c]; d += v * wd[c]; }
  asrc[i] = s; adst[i] = d;
}

__global__ void edge_pass1(const int* __restrict__ src, const int* __restrict__ dst,
                           const float* __restrict__ ea, const float* __restrict__ lattr,
                           const float* __restrict__ we,
                           const float* __restrict__ asrc, const float* __restrict__ adst,
                           float* __restrict__ abuf, float* __restrict__ mbuf) {
  int e = blockIdx.x * blockDim.x + threadIdx.x;
  const int E2 = kE + kN;
  if (e >= E2) return;
  int s, t; const float* ap;
  if (e < kE) { s = src[e]; t = dst[e]; ap = ea + (size_t)e * 3; }
  else        { s = t = e - kE;          ap = lattr + (size_t)(e - kE) * 3; }
  float f0 = ap[0], f1 = ap[1], f2 = ap[2];
  for (int h = 0; h < kH; ++h) {
    float aeh = f0 * we[0 * kH + h] + f1 * we[1 * kH + h] + f2 * we[2 * kH + h];
    float a = asrc[s * kH + h] + adst[t * kH + h] + aeh;
    a = (a > 0.0f) ? a : 0.2f * a;                    // leaky_relu(0.2)
    abuf[(size_t)e * kH + h] = a;
    atomicMaxF(&mbuf[t * kH + h], a);
  }
}

__global__ void edge_pass2(const int* __restrict__ dst,
                           float* __restrict__ abuf, const float* __restrict__ mbuf,
                           float* __restrict__ zbuf) {
  int e = blockIdx.x * blockDim.x + threadIdx.x;
  const int E2 = kE + kN;
  if (e >= E2) return;
  int t = (e < kE) ? dst[e] : (e - kE);
  for (int h = 0; h < kH; ++h) {
    float w = expf(abuf[(size_t)e * kH + h] - mbuf[t * kH + h]);
    abuf[(size_t)e * kH + h] = w;
    atomicAdd(&zbuf[t * kH + h], w);
  }
}

__global__ void edge_pass3(const int* __restrict__ src, const int* __restrict__ dst,
                           const float* __restrict__ hp, const float* __restrict__ wbuf,
                           const float* __restrict__ zbuf, float* __restrict__ outacc) {
  long long idx = (long long)blockIdx.x * blockDim.x + threadIdx.x;   // e*64 + c
  const long long total = (long long)(kE + kN) * kC;
  if (idx >= total) return;
  int e = (int)(idx >> 6), c = (int)(idx & 63);
  int s, t;
  if (e < kE) { s = src[e]; t = dst[e]; } else { s = t = e - kE; }
  for (int h = 0; h < kH; ++h) {
    float alpha = wbuf[(size_t)e * kH + h] / (zbuf[t * kH + h] + 1e-16f);
    atomicAdd(&outacc[(size_t)t * kHP + h * kC + c],
              hp[(size_t)s * kHP + h * kC + c] * alpha);
  }
}

__global__ void head_mean_gelu(const float* __restrict__ outacc, const float* __restrict__ bias,
                               float* __restrict__ hcol) {
  int i = blockIdx.x * blockDim.x + threadIdx.x;   // n*64 + c
  if (i >= kN * kC) return;
  int n = i >> 6, c = i & 63;
  float s = 0.0f;
  for (int h = 0; h < kH; ++h) s += outacc[(size_t)n * kHP + h * kC + c];
  float v = s * (1.0f / (float)kH) + bias[c];
  hcol[(size_t)n * kLD + c] = gelu_exact(v);
}

// Cw = 1<<cshift (64 or 512)
__global__ void bn_reduce(const float* __restrict__ X, int ld, int cshift,
                          float* __restrict__ sum, float* __restrict__ sumsq) {
  __shared__ float ssum[512];
  __shared__ float ssq[512];
  const int Cw = 1 << cshift;
  const int cmask = Cw - 1;
  for (int c = threadIdx.x; c < Cw; c += blockDim.x) { ssum[c] = 0.0f; ssq[c] = 0.0f; }
  __syncthreads();
  int r0 = blockIdx.x * 64;
  for (int i = threadIdx.x; i < (64 << cshift); i += blockDim.x) {
    int r = r0 + (i >> cshift), c = i & cmask;
    if (r < kN) {
      float v = X[(size_t)r * ld + c];
      atomicAdd(&ssum[c], v);
      atomicAdd(&ssq[c], v * v);
    }
  }
  __syncthreads();
  for (int c = threadIdx.x; c < Cw; c += blockDim.x) {
    atomicAdd(&sum[c], ssum[c]);
    atomicAdd(&sumsq[c], ssq[c]);
  }
}

__global__ void bn_apply(float* __restrict__ X, int ld, int cshift,
                         const float* __restrict__ sum, const float* __restrict__ sumsq,
                         const float* __restrict__ g, const float* __restrict__ b) {
  long long i = (long long)blockIdx.x * blockDim.x + threadIdx.x;   // n*Cw + c
  if (i >= ((long long)kN << cshift)) return;
  int n = (int)(i >> cshift), c = (int)(i & ((1 << cshift) - 1));
  float mu  = sum[c] * (1.0f / (float)kN);
  float var = sumsq[c] * (1.0f / (float)kN) - mu * mu;
  float v = X[(size_t)n * ld + c];
  X[(size_t)n * ld + c] = (v - mu) * rsqrtf(var + kEPS) * g[c] + b[c];
}

__global__ void gelu_inplace(float* __restrict__ X, long long n) {
  long long i = (long long)blockIdx.x * blockDim.x + threadIdx.x;
  if (i < n) X[i] = gelu_exact(X[i]);
}

__global__ void pool_cnt(const int* __restrict__ batch, float* __restrict__ cnt) {
  int n = blockIdx.x * blockDim.x + threadIdx.x;
  if (n < kN) atomicAdd(&cnt[batch[n]], 1.0f);
}

__global__ void pool_scatter(const float* __restrict__ y, const int* __restrict__ batch,
                             float* __restrict__ pooled) {
  long long i = (long long)blockIdx.x * blockDim.x + threadIdx.x;   // n*512 + j
  if (i >= (long long)kN * kOUT) return;
  int n = (int)(i >> 9), j = (int)(i & 511);
  atomicAdd(&pooled[(size_t)batch[n] * kOUT + j], y[(size_t)n * kOUT + j]);
}

__global__ void pool_final(float* __restrict__ pooled, const float* __restrict__ cnt) {
  int i = blockIdx.x * blockDim.x + threadIdx.x;   // g*512 + j
  if (i >= kG * kOUT) return;
  pooled[i] /= fmaxf(cnt[i >> 9], 1.0f);
}

// ---------------------------------------------------------------------------
extern "C" void kernel_launch(void* const* d_in, const int* in_sizes, int n_in,
                              void* d_out, int out_size, void* d_ws, size_t ws_size,
                              hipStream_t stream) {
  (void)in_sizes; (void)n_in; (void)out_size; (void)ws_size;
  const float* x     = (const float*)d_in[0];
  const int*   ei    = (const int*)d_in[1];
  const float* eattr = (const float*)d_in[2];
  const int*   batch = (const int*)d_in[3];
  const float* emb_w = (const float*)d_in[4];
  const float* emb_b = (const float*)d_in[5];
  const float* fin_w = (const float*)d_in[6 + 8 * kNL + 0];
  const float* fin_b = (const float*)d_in[6 + 8 * kNL + 1];
  const float* fbn_g = (const float*)d_in[6 + 8 * kNL + 2];
  const float* fbn_b = (const float*)d_in[6 + 8 * kNL + 3];
  const int* src = ei;
  const int* dst = ei + kE;
  float* pooled = (float*)d_out;

  // ---- workspace carve ----
  char* ws = (char*)d_ws;
  size_t off = 0;
  auto carve = [&](size_t bytes) -> char* {
    char* p = ws + off;
    off = (off + bytes + 255) & ~(size_t)255;
    return p;
  };
  float* h      = (float*)carve((size_t)kN * kLD * 4);
  float* hp     = (float*)carve((size_t)kN * kHP * 4);   // also reused as y[N,512]
  float* outacc = (float*)carve((size_t)kN * kHP * 4);
  float* asrc   = (float*)carve((size_t)kN * kH * 4);
  float* adst   = (float*)carve((size_t)kN * kH * 4);
  float* mbuf   = (float*)carve((size_t)kN * kH * 4);
  float* zbuf   = (float*)carve((size_t)kN * kH * 4);
  float* abuf   = (float*)carve((size_t)(kE + kN) * kH * 4);
  float* deg    = (float*)carve((size_t)kN * 4);
  float* lattr  = (float*)carve((size_t)kN * 3 * 4);
  float* webuf  = (float*)carve(3 * kH * 4);
  float* bnsum  = (float*)carve(512 * 4);
  float* bnsq   = (float*)carve(512 * 4);
  float* cnt    = (float*)carve((size_t)kG * 4);

  auto cdiv = [](long long a, long long b) { return (int)((a + b - 1) / b); };
  const int E2 = kE + kN;

  // ---- embedding ----
  embed_kernel<<<cdiv((long long)kN * kEMB, 256), 256, 0, stream>>>(x, emb_w, emb_b, h);

  // ---- self-loop attr (mean of incoming) : once ----
  hipMemsetAsync(deg, 0, (size_t)kN * 4, stream);
  hipMemsetAsync(lattr, 0, (size_t)kN * 3 * 4, stream);
  deg_loop_scatter<<<cdiv(kE, 256), 256, 0, stream>>>(dst, eattr, deg, lattr);
  loop_finalize<<<cdiv((long long)kN * 3, 256), 256, 0, stream>>>(lattr, deg);

  // ---- GAT layers ----
  int d = kEMB;
  for (int l = 0; l < kNL; ++l) {
    const float* lw    = (const float*)d_in[6 + 8 * l + 0];
    const float* at_s  = (const float*)d_in[6 + 8 * l + 1];
    const float* at_d  = (const float*)d_in[6 + 8 * l + 2];
    const float* lin_e = (const float*)d_in[6 + 8 * l + 3];
    const float* at_e  = (const float*)d_in[6 + 8 * l + 4];
    const float* bias  = (const float*)d_in[6 + 8 * l + 5];
    const float* bn_g  = (const float*)d_in[6 + 8 * l + 6];
    const float* bn_b  = (const float*)d_in[6 + 8 * l + 7];

    edge_weight_kernel<<<1, 32, 0, stream>>>(lin_e, at_e, webuf);

    // hp[N,640] = h[:, :d] @ lw  (WMMA)
    gemm_bf16_wmma<<<dim3(cdiv(kN, TM), kHP / TN), 256, 0, stream>>>(
        h, kLD, lw, kHP, nullptr, hp, kHP, kN, d);

    att_node_kernel<<<cdiv((long long)kN * kH, 256), 256, 0, stream>>>(hp, at_s, at_d, asrc, adst);

    fill_f32<<<cdiv((long long)kN * kH, 256), 256, 0, stream>>>(mbuf, -1e30f, (long long)kN * kH);
    hipMemsetAsync(zbuf, 0, (size_t)kN * kH * 4, stream);

    edge_pass1<<<cdiv(E2, 256), 256, 0, stream>>>(src, dst, eattr, lattr, webuf,
                                                  asrc, adst, abuf, mbuf);
    edge_pass2<<<cdiv(E2, 256), 256, 0, stream>>>(dst, abuf, mbuf, zbuf);

    hipMemsetAsync(outacc, 0, (size_t)kN * kHP * 4, stream);
    edge_pass3<<<cdiv((long long)E2 * kC, 256), 256, 0, stream>>>(src, dst, hp, abuf, zbuf, outacc);

    float* hcol = h + d;   // write new 64 columns at offset d (row stride kLD)
    head_mean_gelu<<<cdiv((long long)kN * kC, 256), 256, 0, stream>>>(outacc, bias, hcol);

    hipMemsetAsync(bnsum, 0, 512 * 4, stream);
    hipMemsetAsync(bnsq, 0, 512 * 4, stream);
    bn_reduce<<<cdiv(kN, 64), 256, 0, stream>>>(hcol, kLD, 6, bnsum, bnsq);
    bn_apply<<<cdiv((long long)kN * kC, 256), 256, 0, stream>>>(hcol, kLD, 6, bnsum, bnsq, bn_g, bn_b);

    d += kC;
  }

  // ---- final linear (WMMA) + GELU + BN + mean pool ----
  float* y = hp;   // reuse [N,640] buffer for [N,512]
  gemm_bf16_wmma<<<dim3(cdiv(kN, TM), kOUT / TN), 256, 0, stream>>>(
      h, kLD, fin_w, kOUT, fin_b, y, kOUT, kN, kLD);
  gelu_inplace<<<cdiv((long long)kN * kOUT, 256), 256, 0, stream>>>(y, (long long)kN * kOUT);

  hipMemsetAsync(bnsum, 0, 512 * 4, stream);
  hipMemsetAsync(bnsq, 0, 512 * 4, stream);
  bn_reduce<<<cdiv(kN, 64), 256, 0, stream>>>(y, kOUT, 9, bnsum, bnsq);
  bn_apply<<<cdiv((long long)kN * kOUT, 256), 256, 0, stream>>>(y, kOUT, 9, bnsum, bnsq, fbn_g, fbn_b);

  hipMemsetAsync(pooled, 0, (size_t)kG * kOUT * 4, stream);
  hipMemsetAsync(cnt, 0, (size_t)kG * 4, stream);
  pool_cnt<<<cdiv(kN, 256), 256, 0, stream>>>(batch, cnt);
  pool_scatter<<<cdiv((long long)kN * kOUT, 256), 256, 0, stream>>>(y, batch, pooled);
  pool_final<<<cdiv((long long)kG * kOUT, 256), 256, 0, stream>>>(pooled, cnt);
}
